// KGAT_81612968559390
// MI455X (gfx1250) — matrix operand order repaired and checked by reading, not compiled, not run on hardware
//
#include <hip/hip_runtime.h>
#include <math.h>

// ---------------------------------------------------------------------------
// KGAT forward for MI455X (gfx1250, wave32, WMMA).
//   - fused emb @ W_R[0..15] transform: one wave holds the A fragments in
//     VGPRs and loops all 16 relations (128 v_wmma per wave), f16 output
//   - edge attention (f16 gathers) + segment softmax (sorted dst, atomics)
//   - GCN chain c1..c4 (4 aggregations, WMMA GEMMs w/ fused bias+LeakyReLU)
//   - linear tails t2,t3,t4,u3,u4,v4 (dedup of the reference's index loop)
//   - L2-normalized concat into out[4][N][136]
// ---------------------------------------------------------------------------

typedef _Float16 v2h  __attribute__((ext_vector_type(2)));
typedef _Float16 v8h  __attribute__((ext_vector_type(8)));
typedef _Float16 v16h __attribute__((ext_vector_type(16)));
typedef float    v8f  __attribute__((ext_vector_type(8)));

static constexpr int N_NODES = 100000;
static constexpr int N_REL   = 16;
static constexpr int N_EDGES = 1600000;
static constexpr int D       = 64;
static constexpr float NEG_SLOPE = 0.01f;
static constexpr float EPS = 1e-12f;

// ---------------------------------------------------------------------------
// Pack an fp32 weight matrix W (Kin x Nout, row-major) into the CDNA5 16-bit
// B-fragment order (32-K-step x 16-col tiles), zero-padded to Kp x 16*ctiles.
// Fragment layout matching A: element j<8 -> K=8h+j, j>=8 -> K=16+8h+(j-8)
// with h = lane/16; N = lane%16.
// ---------------------------------------------------------------------------
__global__ void pack_b_kernel(const float* __restrict__ W, int Kin, int Nout,
                              _Float16* __restrict__ out, int Kp, int ctiles) {
  int total = (Kp / 32) * ctiles * 512;
  int idx = blockIdx.x * blockDim.x + threadIdx.x;
  if (idx >= total) return;
  int j    = idx & 15;
  int lane = (idx >> 4) & 31;
  int rest = idx >> 9;               // ks*ctiles + c
  int c    = rest % ctiles;
  int ks   = rest / ctiles;
  int vg = j >> 1, comp = j & 1;
  int K  = ks * 32 + (j & 7) + ((lane & 16) ? 8 : 0) + ((j & 8) ? 16 : 0);
  int Nc = c * 16 + (lane & 15);
  float v = (K < Kin && Nc < Nout) ? W[(size_t)K * Nout + Nc] : 0.0f;
  out[idx] = (_Float16)v;
}

// Convert fp32 X (nrows x Kin) [optionally + Y] to f16 row-major nrows x Kp,
// zero-padding K. Used for emb, and for (ego + N_h) before each WMMA GEMM.
__global__ void cvt_a_kernel(const float* __restrict__ X, const float* __restrict__ Y,
                             _Float16* __restrict__ out, int Kin, int Kp, int nrows) {
  size_t i = (size_t)blockIdx.x * blockDim.x + threadIdx.x;
  size_t total = (size_t)nrows * Kp;
  if (i >= total) return;
  int k = (int)(i % Kp);
  size_t row = i / Kp;
  float v = 0.0f;
  if (k < Kin) {
    v = X[row * Kin + k];
    if (Y) v += Y[row * Kin + k];
  }
  out[i] = (_Float16)v;
}

// Load one A fragment (ISA 16-bit A 16x32 layout) for a 16-row tile.
// lane<16: row=lane,   K in {off..off+7, off+16..off+23}, off = 32*ks
// lane>=16: row=lane-16, same pattern with off += 8
__device__ inline v16h load_a_frag(const _Float16* __restrict__ arow, int off) {
  v8h alo = *(const v8h*)(arow + off);
  v8h ahi = *(const v8h*)(arow + off + 16);
  v16h a;
#pragma unroll
  for (int i = 0; i < 8; ++i) { a[i] = alo[i]; a[i + 8] = ahi[i]; }
  return a;
}

// ---------------------------------------------------------------------------
// Fused transform: all_mul16[n][r][:] = f16( emb[n] @ W_R[r] ), r = 0..15.
// One wave per 16-row tile; A fragments (2 k-steps) live in VGPRs across the
// whole relation loop -> emb read exactly once (16x less A traffic than 16
// separate GEMMs). 128 v_wmma_f32_16x16x32_f16 per wave.
// ---------------------------------------------------------------------------
__global__ __launch_bounds__(256)
void transform_wmma_kernel(const _Float16* __restrict__ A16,
                           const _Float16* __restrict__ Bpack,  // 16 rel x 4096
                           _Float16* __restrict__ out, int nrows) {
  const int lane = threadIdx.x & 31;
  const int wv   = threadIdx.x >> 5;
  const int tile = blockIdx.x * 8 + wv;
  if (tile * 16 >= nrows) return;   // wave-uniform: EXEC all-ones for WMMA

  const _Float16* arow = A16 + (size_t)(tile * 16 + (lane & 15)) * 64;
  const int koff = (lane & 16) ? 8 : 0;
  v16h a0 = load_a_frag(arow, koff);        // K  0..31
  v16h a1 = load_a_frag(arow, 32 + koff);   // K 32..63

  const int half = (lane & 16) ? 8 : 0;
  const int coll = lane & 15;

  for (int r = 0; r < N_REL; ++r) {
    v8f acc[4] = {};
#pragma unroll
    for (int c = 0; c < 4; ++c) {
      v16h b0 = *(const v16h*)(Bpack + (size_t)((r * 2 + 0) * 4 + c) * 512 + lane * 16);
      acc[c] = __builtin_amdgcn_wmma_f32_16x16x32_f16(
          false, a0, false, b0, (short)0, acc[c], false, false);
      v16h b1 = *(const v16h*)(Bpack + (size_t)((r * 2 + 1) * 4 + c) * 512 + lane * 16);
      acc[c] = __builtin_amdgcn_wmma_f32_16x16x32_f16(
          false, a1, false, b1, (short)0, acc[c], false, false);
    }
    // C layout: VGPR v -> rows v (lanes 0-15) / v+8 (lanes 16-31), col=lane%16
#pragma unroll
    for (int c = 0; c < 4; ++c)
#pragma unroll
      for (int v = 0; v < 8; ++v)
        out[((size_t)(tile * 16 + v + half) * N_REL + r) * D + c * 16 + coll] =
            (_Float16)acc[c][v];
  }
}

// ---------------------------------------------------------------------------
// Generic WMMA GEMM for the MLP layers: C = A16(nrows x Kp) @ Bpack, fp32 out.
// Optional fused  y = leakyrelu(x + bias[col]).
// ---------------------------------------------------------------------------
template <int KSTEPS, int CTILES>
__global__ __launch_bounds__(256)
void gemm_wmma_kernel(const _Float16* __restrict__ A16,
                      const _Float16* __restrict__ Bpack,
                      float* __restrict__ C, int ldc, int ncols,
                      const float* __restrict__ bias, int do_lrelu, int nrows) {
  constexpr int Kp = KSTEPS * 32;
  const int lane = threadIdx.x & 31;
  const int wv   = threadIdx.x >> 5;
  const int tile = blockIdx.x * 8 + wv;
  if (tile * 16 >= nrows) return;   // wave-uniform

  v8f acc[CTILES] = {};
  const _Float16* arow = A16 + (size_t)(tile * 16 + (lane & 15)) * Kp;
  const int koff = (lane & 16) ? 8 : 0;

#pragma unroll
  for (int ks = 0; ks < KSTEPS; ++ks) {
    v16h a = load_a_frag(arow, ks * 32 + koff);
#pragma unroll
    for (int c = 0; c < CTILES; ++c) {
      v16h b = *(const v16h*)(Bpack + (size_t)(ks * CTILES + c) * 512 + lane * 16);
      acc[c] = __builtin_amdgcn_wmma_f32_16x16x32_f16(
          false, a, false, b, (short)0, acc[c], false, false);
    }
  }

  const int half = (lane & 16) ? 8 : 0;
  const int coll = lane & 15;
#pragma unroll
  for (int c = 0; c < CTILES; ++c) {
    int col = c * 16 + coll;
    if (col < ncols) {
      float bv = bias ? bias[col] : 0.0f;
#pragma unroll
      for (int v = 0; v < 8; ++v) {
        float x = acc[c][v] + bv;
        if (do_lrelu) x = (x >= 0.0f) ? x : NEG_SLOPE * x;
        C[(size_t)(tile * 16 + v + half) * ldc + col] = x;
      }
    }
  }
}

// ---------------------------------------------------------------------------
// Edge attention:  att_raw[e] = sum_d t[d] * tanh(h[d] + rel[d])
// all_mul16 is f16 (halves the dominant gather traffic). One wave32 per edge,
// lanes cover 2 features each; wave32 __shfl_xor tree reduction.
// ---------------------------------------------------------------------------
__global__ void att_edge_kernel(const _Float16* __restrict__ all_mul16,
                                const float* __restrict__ rel,
                                const int* __restrict__ src, const int* __restrict__ dst,
                                const int* __restrict__ ety,
                                float* __restrict__ att_raw, int nedges) {
  int e = blockIdx.x * 8 + (threadIdx.x >> 5);
  if (e >= nedges) return;
  int lane = threadIdx.x & 31;
  int r = ety[e];
  v2h t  = *(const v2h*)(all_mul16 + ((size_t)src[e] * N_REL + r) * D + lane * 2);
  v2h h  = *(const v2h*)(all_mul16 + ((size_t)dst[e] * N_REL + r) * D + lane * 2);
  const float2 rr = *(const float2*)(rel + (size_t)r * D + lane * 2);
  float acc = (float)t[0] * tanhf((float)h[0] + rr.x)
            + (float)t[1] * tanhf((float)h[1] + rr.y);
#pragma unroll
  for (int off = 16; off > 0; off >>= 1) acc += __shfl_xor(acc, off, 32);
  if (lane == 0) att_raw[e] = acc;
}

__global__ void fill_kernel(float* __restrict__ p, float v, size_t n) {
  size_t i = (size_t)blockIdx.x * blockDim.x + threadIdx.x;
  if (i < n) p[i] = v;
}

// float atomic max via int/uint bit trick (monotone orderings)
__device__ inline void atomicMaxF(float* addr, float val) {
  if (val >= 0.0f) atomicMax((int*)addr, __float_as_int(val));
  else             atomicMin((unsigned int*)addr, __float_as_uint(val));
}

__global__ void edge_max_kernel(const float* __restrict__ att_raw,
                                const int* __restrict__ dst,
                                float* __restrict__ m, int nedges) {
  int e = blockIdx.x * blockDim.x + threadIdx.x;
  if (e >= nedges) return;
  atomicMaxF(&m[dst[e]], att_raw[e]);
}

__global__ void edge_exp_kernel(const float* __restrict__ att_raw,
                                const int* __restrict__ dst,
                                const float* __restrict__ m,
                                float* __restrict__ eb, float* __restrict__ s, int nedges) {
  int e = blockIdx.x * blockDim.x + threadIdx.x;
  if (e >= nedges) return;
  float ex = expf(att_raw[e] - m[dst[e]]);
  eb[e] = ex;
  atomicAdd(&s[dst[e]], ex);
}

__global__ void edge_norm_kernel(const float* __restrict__ eb,
                                 const int* __restrict__ dst,
                                 const float* __restrict__ s,
                                 float* __restrict__ attn, int nedges) {
  int e = blockIdx.x * blockDim.x + threadIdx.x;
  if (e >= nedges) return;
  attn[e] = eb[e] / s[dst[e]];
}

// N_h[dst] += att[e] * ego[src]   (wave per edge, lanes stride the feature dim)
__global__ void aggregate_kernel(const float* __restrict__ att,
                                 const float* __restrict__ ego,
                                 const int* __restrict__ src, const int* __restrict__ dst,
                                 float* __restrict__ Nh, int dim, int nedges) {
  int e = blockIdx.x * 8 + (threadIdx.x >> 5);
  if (e >= nedges) return;
  int lane = threadIdx.x & 31;
  float a = att[e];
  size_t sb = (size_t)src[e] * dim;
  size_t db = (size_t)dst[e] * dim;
  for (int d = lane; d < dim; d += 32)
    atomicAdd(&Nh[db + d], a * ego[sb + d]);
}

// out[obase + n*136 + 0..63] = emb[n][0..63]  (all_embed[0] is un-normalized)
__global__ void copy_emb_kernel(const float* __restrict__ emb,
                                float* __restrict__ out, size_t obase, int nrows) {
  size_t i = (size_t)blockIdx.x * blockDim.x + threadIdx.x;
  if (i >= (size_t)nrows * 64) return;
  size_t n = i >> 6;
  int d = (int)(i & 63);
  out[obase + n * 136 + d] = emb[i];
}

// Row-wise L2 normalize (dim <= 32) into output slice at column coloff.
__global__ void store_norm_kernel(const float* __restrict__ X, int dim,
                                  float* __restrict__ out, size_t obase,
                                  int coloff, int nrows) {
  int n = blockIdx.x * blockDim.x + threadIdx.x;
  if (n >= nrows) return;
  float buf[32];
  float ss = 0.0f;
  for (int d = 0; d < dim; ++d) {
    float x = X[(size_t)n * dim + d];
    buf[d] = x;
    ss += x * x;
  }
  float inv = 1.0f / fmaxf(sqrtf(ss), EPS);
  for (int d = 0; d < dim; ++d)
    out[obase + (size_t)n * 136 + coloff + d] = buf[d] * inv;
}

// ---------------------------------------------------------------------------
extern "C" void kernel_launch(void* const* d_in, const int* in_sizes, int n_in,
                              void* d_out, int out_size, void* d_ws, size_t ws_size,
                              hipStream_t stream) {
  (void)in_sizes; (void)n_in; (void)out_size; (void)ws_size;
  const float* emb = (const float*)d_in[0];
  const float* rel = (const float*)d_in[1];
  const float* W_R = (const float*)d_in[2];
  const float* W0  = (const float*)d_in[3];  const float* b0 = (const float*)d_in[4];
  const float* W1  = (const float*)d_in[5];  const float* b1 = (const float*)d_in[6];
  const float* W2  = (const float*)d_in[7];  const float* b2 = (const float*)d_in[8];
  const float* W3  = (const float*)d_in[9];  const float* b3 = (const float*)d_in[10];
  const int* src  = (const int*)d_in[11];
  const int* dst  = (const int*)d_in[12];
  const int* ety  = (const int*)d_in[13];
  float* out = (float*)d_out;

  const int N = N_NODES, E = N_EDGES;

  // ---- workspace carve (256 B aligned) ----
  char* w = (char*)d_ws;
  auto carve = [&](size_t bytes) -> char* {
    char* r = w;
    w += (bytes + 255) & ~(size_t)255;
    return r;
  };
  _Float16* emb16   = (_Float16*)carve((size_t)N * 64 * 2);
  _Float16* BpWR    = (_Float16*)carve((size_t)16 * 4096 * 2);
  _Float16* BpW0    = (_Float16*)carve(2048 * 2);
  _Float16* BpW1    = (_Float16*)carve(512 * 2);
  _Float16* BpW2    = (_Float16*)carve(512 * 2);
  _Float16* BpW3    = (_Float16*)carve(512 * 2);
  _Float16* all_mul16 = (_Float16*)carve((size_t)N * N_REL * D * 2);
  float* att_raw   = (float*)carve((size_t)E * 4);
  float* eb        = (float*)carve((size_t)E * 4);
  float* attn      = (float*)carve((size_t)E * 4);
  float* mbuf      = (float*)carve((size_t)N * 4);
  float* sbuf      = (float*)carve((size_t)N * 4);
  float* Nh        = (float*)carve((size_t)N * 64 * 4);
  _Float16* A16buf = (_Float16*)carve((size_t)N * 64 * 2);
  float* c1 = (float*)carve((size_t)N * 32 * 4);
  float* c2 = (float*)carve((size_t)N * 16 * 4);
  float* c3 = (float*)carve((size_t)N * 16 * 4);
  float* c4 = (float*)carve((size_t)N * 8 * 4);
  float* t2 = (float*)carve((size_t)N * 16 * 4);
  float* t3 = (float*)carve((size_t)N * 16 * 4);
  float* t4 = (float*)carve((size_t)N * 8 * 4);
  float* u3 = (float*)carve((size_t)N * 16 * 4);
  float* u4 = (float*)carve((size_t)N * 8 * 4);
  float* v4 = (float*)carve((size_t)N * 8 * 4);

  const int B = 256;
  const int gemm_blocks = (N / 16 + 7) / 8;          // 782 (8 waves/block)
  const int edge_wave_blocks = E / 8;                // 200000 (wave per edge)
  const int edge_thr_blocks  = (E + B - 1) / B;      // 6250

  // ---- pack weights into WMMA B-fragment order ----
  for (int r = 0; r < N_REL; ++r)
    pack_b_kernel<<<16, B, 0, stream>>>(W_R + (size_t)r * D * D, 64, 64,
                                        BpWR + (size_t)r * 4096, 64, 4);
  pack_b_kernel<<<8, B, 0, stream>>>(W0, 64, 32, BpW0, 64, 2);
  pack_b_kernel<<<2, B, 0, stream>>>(W1, 32, 16, BpW1, 32, 1);
  pack_b_kernel<<<2, B, 0, stream>>>(W2, 16, 16, BpW2, 32, 1);
  pack_b_kernel<<<2, B, 0, stream>>>(W3, 16,  8, BpW3, 32, 1);

  // ---- fused transform: all 16 relations, A kept in VGPRs ----
  cvt_a_kernel<<<(int)(((size_t)N * 64 + B - 1) / B), B, 0, stream>>>(
      emb, nullptr, emb16, 64, 64, N);
  transform_wmma_kernel<<<gemm_blocks, B, 0, stream>>>(emb16, BpWR, all_mul16, N);

  // ---- edge attention + segment softmax over dst ----
  att_edge_kernel<<<edge_wave_blocks, B, 0, stream>>>(all_mul16, rel, src, dst, ety,
                                                      att_raw, E);
  fill_kernel<<<(N + B - 1) / B, B, 0, stream>>>(mbuf, -INFINITY, (size_t)N);
  fill_kernel<<<(N + B - 1) / B, B, 0, stream>>>(sbuf, 0.0f, (size_t)N);
  edge_max_kernel<<<edge_thr_blocks, B, 0, stream>>>(att_raw, dst, mbuf, E);
  edge_exp_kernel<<<edge_thr_blocks, B, 0, stream>>>(att_raw, dst, mbuf, eb, sbuf, E);
  edge_norm_kernel<<<edge_thr_blocks, B, 0, stream>>>(eb, dst, sbuf, attn, E);

  // ---- GCN chain: c_{i+1} = lrelu((c_i + agg(c_i)) @ W_i + b_i) ----
  struct Step { const float* ego; int dim; _Float16* Bp; float* Cout;
                int dout; const float* bias; };
  Step steps[4] = {
    { emb, 64, BpW0, c1, 32, b0 },
    { c1,  32, BpW1, c2, 16, b1 },
    { c2,  16, BpW2, c3, 16, b2 },
    { c3,  16, BpW3, c4,  8, b3 },
  };
  for (int i = 0; i < 4; ++i) {
    const Step& st = steps[i];
    size_t nh_elems = (size_t)N * st.dim;
    fill_kernel<<<(int)((nh_elems + B - 1) / B), B, 0, stream>>>(Nh, 0.0f, nh_elems);
    aggregate_kernel<<<edge_wave_blocks, B, 0, stream>>>(attn, st.ego, src, dst,
                                                         Nh, st.dim, E);
    int Kp = (st.dim == 64) ? 64 : 32;
    cvt_a_kernel<<<(int)(((size_t)N * Kp + B - 1) / B), B, 0, stream>>>(
        st.ego, Nh, A16buf, st.dim, Kp, N);
    if (i == 0)
      gemm_wmma_kernel<2, 2><<<gemm_blocks, B, 0, stream>>>(
          A16buf, st.Bp, st.Cout, st.dout, st.dout, st.bias, 1, N);
    else
      gemm_wmma_kernel<1, 1><<<gemm_blocks, B, 0, stream>>>(
          A16buf, st.Bp, st.Cout, st.dout, st.dout, st.bias, 1, N);
  }

  // ---- linear tails: lrelu(x @ W + b), no aggregation ----
  struct Lin { const float* X; int dim; _Float16* Bp; float* Cout; int dout;
               const float* bias; };
  Lin lins[6] = {
    { c1, 32, BpW1, t2, 16, b1 },
    { t2, 16, BpW2, t3, 16, b2 },
    { t3, 16, BpW3, t4,  8, b3 },
    { c2, 16, BpW2, u3, 16, b2 },
    { u3, 16, BpW3, u4,  8, b3 },
    { c3, 16, BpW3, v4,  8, b3 },
  };
  for (int i = 0; i < 6; ++i) {
    const Lin& ln = lins[i];
    cvt_a_kernel<<<(int)(((size_t)N * 32 + B - 1) / B), B, 0, stream>>>(
        ln.X, nullptr, A16buf, ln.dim, 32, N);
    gemm_wmma_kernel<1, 1><<<gemm_blocks, B, 0, stream>>>(
        A16buf, ln.Bp, ln.Cout, ln.dout, ln.dout, ln.bias, 1, N);
  }

  // ---- assemble out[4][N][136]: [emb | norm blocks @ cols 64,96,112,128] ----
  size_t stride = (size_t)N * 136;
  for (int i = 0; i < 4; ++i)
    copy_emb_kernel<<<(int)(((size_t)N * 64 + B - 1) / B), B, 0, stream>>>(
        emb, out, (size_t)i * stride, N);

  auto SN = [&](const float* X, int dim, int slot, int coloff) {
    store_norm_kernel<<<(N + B - 1) / B, B, 0, stream>>>(
        X, dim, out, (size_t)slot * stride, coloff, N);
  };
  // index=1
  SN(c1, 32, 0, 64); SN(t2, 16, 0, 96); SN(t3, 16, 0, 112); SN(t4, 8, 0, 128);
  // index=2
  SN(c1, 32, 1, 64); SN(c2, 16, 1, 96); SN(u3, 16, 1, 112); SN(u4, 8, 1, 128);
  // index=3
  SN(c1, 32, 2, 64); SN(c2, 16, 2, 96); SN(c3, 16, 2, 112); SN(v4, 8, 2, 128);
  // index=4
  SN(c1, 32, 3, 64); SN(c2, 16, 3, 96); SN(c3, 16, 3, 112); SN(c4, 8, 3, 128);
}
